// HierarchicalDynamicsModel_37220186587417
// MI455X (gfx1250) — matrix operand order repaired and checked by reading, not compiled, not run on hardware
//
#include <hip/hip_runtime.h>

#define DFEAT 128

typedef __attribute__((ext_vector_type(16))) __bf16 v16bf;
typedef __attribute__((ext_vector_type(2)))  __bf16 v2bf;
typedef __attribute__((ext_vector_type(8)))  float  v8f;

union BF16x16 { v16bf v; unsigned short u[16]; };

// ---- bf16 conversion helpers (prefer v_cvt_pk_bf16_f32) ----------------------
__device__ __forceinline__ unsigned short f2bf_manual(float f) {
  unsigned int x = __float_as_uint(f);
  x += 0x7FFFu + ((x >> 16) & 1u);   // round-to-nearest-even
  return (unsigned short)(x >> 16);
}
__device__ __forceinline__ unsigned cvt2_bf16(float lo, float hi) {
#if __has_builtin(__builtin_amdgcn_cvt_pk_bf16_f32)
  union { v2bf v; unsigned u; } cv;
  cv.v = __builtin_amdgcn_cvt_pk_bf16_f32(lo, hi);
  return cv.u;
#else
  return (unsigned)f2bf_manual(lo) | ((unsigned)f2bf_manual(hi) << 16);
#endif
}
__device__ __forceinline__ unsigned short f2bf16(float f) {
#if __has_builtin(__builtin_amdgcn_cvt_pk_bf16_f32)
  return (unsigned short)(cvt2_bf16(f, 0.0f) & 0xFFFFu);
#else
  return f2bf_manual(f);
#endif
}

__device__ __forceinline__ v8f wmma_bf16(v16bf a, v16bf b, v8f c) {
  // v_wmma_f32_16x16x32_bf16: D = A(16x32) * B(32x16) + C(16x16 f32)
  return __builtin_amdgcn_wmma_f32_16x16x32_bf16(false, a, false, b, (short)0, c, false, false);
}

// ---- CDNA5 async global->LDS copy (ASYNCcnt path), ISA cdna5 ch.10/15.18 ----
typedef __attribute__((address_space(3))) const void lds_cvoid_t;

__device__ __forceinline__ unsigned lds_off_of(const void* p) {
  return (unsigned)(unsigned long long)(lds_cvoid_t*)p;
}
__device__ __forceinline__ void async_copy_b128(unsigned lds_off, const void* gptr) {
  asm volatile("global_load_async_to_lds_b128 %0, %1, off"
               :
               : "v"(lds_off), "v"((unsigned long long)(uintptr_t)gptr)
               : "memory");
}
__device__ __forceinline__ void wait_async0() {
  asm volatile("s_wait_asynccnt 0x0" ::: "memory");
}
__device__ __forceinline__ void wait_async1() {
  asm volatile("s_wait_asynccnt 0x1" ::: "memory");
}
__device__ __forceinline__ void wait_async2() {
  asm volatile("s_wait_asynccnt 0x2" ::: "memory");
}

// A fragment (16x32 bf16) from LDS, row-major with row_stride (in ushorts).
__device__ __forceinline__ v16bf load_A_frag(const unsigned short* lds, int row_stride,
                                             int k0, int lane) {
  BF16x16 a;
  const int m  = lane & 15;
  const int kb = (lane >> 4) * 8;
  const unsigned short* p = lds + m * row_stride + k0;
#pragma unroll
  for (int j = 0; j < 8; ++j) {
    const int k = (j < 4) ? (kb + 2 * j) : (16 + kb + 2 * (j - 4));
    a.u[2 * j]     = p[k];
    a.u[2 * j + 1] = p[k + 1];
  }
  return a.v;
}

// B fragment (32x16 bf16) from a global weight matrix stored [K x 128] row-major.
__device__ __forceinline__ v16bf load_B_frag(const unsigned short* __restrict__ W,
                                             int k0, int n0, int lane) {
  BF16x16 b;
  const int n  = n0 + (lane & 15);
  const int kb = k0 + (lane >> 4) * 8;
  const unsigned short* p = W + (size_t)kb * DFEAT + n;
#pragma unroll
  for (int j = 0; j < 8; ++j) {
    const int rk = (j < 4) ? (2 * j) : (8 + 2 * j);
    b.u[2 * j]     = p[(size_t)rk * DFEAT];
    b.u[2 * j + 1] = p[(size_t)(rk + 1) * DFEAT];
  }
  return b.v;
}

__global__ void f32_to_bf16_kernel(const float* __restrict__ src,
                                   unsigned short* __restrict__ dst, int n) {
  int i4 = (blockIdx.x * blockDim.x + threadIdx.x) * 4;
  if (i4 + 3 < n) {
    float4 v = *(const float4*)(src + i4);
    uint2 r;
    r.x = cvt2_bf16(v.x, v.y);
    r.y = cvt2_bf16(v.z, v.w);
    *(uint2*)(dst + i4) = r;
  } else {
    for (int i = i4; i < n; ++i) dst[i] = f2bf16(src[i]);
  }
}

// ---------------------------------------------------------------------------
// Edge MLP (double-buffered async pipeline):
//   x=[node[s],node[r]] (256 bf16), h=relu(x@W1+b1), y=h@W2+b2,
//   atomic scatter-add y into msg[recv].  8 waves, one 16-col N-tile each.
// ---------------------------------------------------------------------------
#define SA_STRIDE 264   // 256 + 8 pad (ushorts); 528B rows, 16B aligned
#define SH_STRIDE 136   // 128 + 8 pad (ushorts); 272B rows, 16B aligned

__global__ __launch_bounds__(256) void edge_mlp_kernel(
    const unsigned short* __restrict__ nodebf, const int* __restrict__ edges,
    const unsigned short* __restrict__ W1, const float* __restrict__ b1,
    const unsigned short* __restrict__ W2, const float* __restrict__ b2,
    float* __restrict__ msg, int n_tiles) {
  __shared__ unsigned int sA_u[2][16 * SA_STRIDE / 2];
  __shared__ unsigned int sH_u[16 * SH_STRIDE / 2];
  unsigned short* sH = (unsigned short*)sH_u;

  const int t     = threadIdx.x;
  const int lane  = t & 31;
  const int wave  = t >> 5;
  const int n0    = wave * 16;
  const int nlane = n0 + (lane & 15);
  const int mhi   = (lane >> 4) * 8;
  const unsigned sA_base[2] = { lds_off_of(&sA_u[0][0]), lds_off_of(&sA_u[1][0]) };

  // per-thread gather coordinates: row gr of tile, 16 bf16 cols at gc0
  const int gr  = t >> 4;
  const int gc0 = (t & 15) * 16;

  v16bf B1[8], B2[4];
#pragma unroll
  for (int kc = 0; kc < 8; ++kc) B1[kc] = load_B_frag(W1, 32 * kc, n0, lane);
#pragma unroll
  for (int kc = 0; kc < 4; ++kc) B2[kc] = load_B_frag(W2, 32 * kc, n0, lane);
  const float bias1 = b1[nlane];
  const float bias2 = b2[nlane];

  auto issue = [&](int tl, unsigned base) {
    const int e = tl * 16 + gr;
    const int s = edges[2 * e];
    const int d = edges[2 * e + 1];
    const unsigned short* g = (gc0 < 128)
        ? (nodebf + (size_t)s * DFEAT + gc0)
        : (nodebf + (size_t)d * DFEAT + (gc0 - 128));
    const unsigned lo = base + (unsigned)(gr * SA_STRIDE + gc0) * 2u;
    async_copy_b128(lo,      g);       // 8 bf16
    async_copy_b128(lo + 16, g + 8);   // 8 bf16
  };

  int tile = blockIdx.x;
  if (tile < n_tiles) issue(tile, sA_base[0]);     // prologue

  int it = 0;
  for (; tile < n_tiles; tile += gridDim.x, ++it) {
    const int buf = it & 1;
    const unsigned short* sA = (const unsigned short*)sA_u[buf];
    const int nxt = tile + gridDim.x;                      // block-uniform
    if (nxt < n_tiles) { issue(nxt, sA_base[buf ^ 1]); wait_async2(); }
    else               { wait_async0(); }
    __syncthreads();   // all waves' tile data resident; prev iter fully done

    v8f c = {0.f, 0.f, 0.f, 0.f, 0.f, 0.f, 0.f, 0.f};
#pragma unroll
    for (int kc = 0; kc < 8; ++kc) {
      v16bf a = load_A_frag(sA, SA_STRIDE, 32 * kc, lane);
      c = wmma_bf16(a, B1[kc], c);
    }
#pragma unroll
    for (int r = 0; r < 8; ++r) {
      float h = fmaxf(c[r] + bias1, 0.0f);
      sH[(r + mhi) * SH_STRIDE + nlane] = f2bf16(h);
    }
    __syncthreads();

    v8f c2 = {0.f, 0.f, 0.f, 0.f, 0.f, 0.f, 0.f, 0.f};
#pragma unroll
    for (int kc = 0; kc < 4; ++kc) {
      v16bf a = load_A_frag(sH, SH_STRIDE, 32 * kc, lane);
      c2 = wmma_bf16(a, B2[kc], c2);
    }
    const int e0 = tile * 16;
#pragma unroll
    for (int r = 0; r < 8; ++r) {
      const int rv = edges[2 * (e0 + r + mhi) + 1];   // L2-hot
      unsafeAtomicAdd(msg + (size_t)rv * DFEAT + nlane, c2[r] + bias2);
    }
  }
}

// ---------------------------------------------------------------------------
// Particle->cluster: h = relu(P @ W_pc + b); scatter-add into c_sum[p2c], counts.
// Double-buffered async pipeline, 1 async B128 per thread per tile.
// ---------------------------------------------------------------------------
__global__ __launch_bounds__(256) void particle_cluster_kernel(
    const unsigned short* __restrict__ nodebf, const int* __restrict__ p2c,
    const unsigned short* __restrict__ W, const float* __restrict__ b,
    float* __restrict__ c_sum, float* __restrict__ counts, int n_tiles) {
  __shared__ unsigned int sA_u[2][16 * SH_STRIDE / 2];

  const int t     = threadIdx.x;
  const int lane  = t & 31;
  const int wave  = t >> 5;
  const int n0    = wave * 16;
  const int nlane = n0 + (lane & 15);
  const int mhi   = (lane >> 4) * 8;
  const unsigned sA_base[2] = { lds_off_of(&sA_u[0][0]), lds_off_of(&sA_u[1][0]) };

  const int gr  = t >> 4;
  const int gc0 = (t & 15) * 8;

  v16bf B[4];
#pragma unroll
  for (int kc = 0; kc < 4; ++kc) B[kc] = load_B_frag(W, 32 * kc, n0, lane);
  const float bias = b[nlane];

  auto issue = [&](int tl, unsigned base) {
    async_copy_b128(base + (unsigned)(gr * SH_STRIDE + gc0) * 2u,
                    nodebf + (size_t)(tl * 16 + gr) * DFEAT + gc0);
  };

  int tile = blockIdx.x;
  if (tile < n_tiles) issue(tile, sA_base[0]);

  int it = 0;
  for (; tile < n_tiles; tile += gridDim.x, ++it) {
    const int buf = it & 1;
    const unsigned short* sA = (const unsigned short*)sA_u[buf];
    const int nxt = tile + gridDim.x;
    if (nxt < n_tiles) { issue(nxt, sA_base[buf ^ 1]); wait_async1(); }
    else               { wait_async0(); }
    __syncthreads();

    const int p0 = tile * 16;
    if (t < 16) unsafeAtomicAdd(counts + p2c[p0 + t], 1.0f);

    v8f c = {0.f, 0.f, 0.f, 0.f, 0.f, 0.f, 0.f, 0.f};
#pragma unroll
    for (int kc = 0; kc < 4; ++kc) {
      v16bf a = load_A_frag(sA, SH_STRIDE, 32 * kc, lane);
      c = wmma_bf16(a, B[kc], c);
    }
#pragma unroll
    for (int r = 0; r < 8; ++r) {
      float h = fmaxf(c[r] + bias, 0.0f);
      const int ci = p2c[p0 + r + mhi];   // L2-hot
      unsafeAtomicAdd(c_sum + (size_t)ci * DFEAT + nlane, h);
    }
  }
}

// cluster_agg = c_sum / max(counts,1), emitted directly as bf16 gather table.
__global__ void mean_div_kernel(const float* __restrict__ c_sum,
                                const float* __restrict__ counts,
                                unsigned short* __restrict__ agg_bf, int n) {
  int i = blockIdx.x * blockDim.x + threadIdx.x;
  if (i < n) agg_bf[i] = f2bf16(c_sum[i] / fmaxf(counts[i >> 7], 1.0f));
}

// ---------------------------------------------------------------------------
// Fused tail: t = relu(cluster_msg[p2c] @ W_cp + b_cp);
//             out = relu([particle_msg, t] @ W_fu + b_fu)
// ---------------------------------------------------------------------------
__global__ __launch_bounds__(256) void fused_out_kernel(
    const float* __restrict__ pmsg, const unsigned short* __restrict__ cmsg_bf,
    const int* __restrict__ p2c,
    const unsigned short* __restrict__ Wcp, const float* __restrict__ bcp,
    const unsigned short* __restrict__ Wfu, const float* __restrict__ bfu,
    float* __restrict__ out, int n_tiles) {
  __shared__ unsigned int sA_u[16 * SH_STRIDE / 2];   // gathered cluster_msg (bf16)
  __shared__ unsigned int sC_u[16 * SA_STRIDE / 2];   // combined [pmsg | t] (bf16)
  unsigned short* sA = (unsigned short*)sA_u;
  unsigned short* sC = (unsigned short*)sC_u;

  const int t     = threadIdx.x;
  const int lane  = t & 31;
  const int wave  = t >> 5;
  const int n0    = wave * 16;
  const int nlane = n0 + (lane & 15);
  const int mhi   = (lane >> 4) * 8;
  const unsigned sA_base = lds_off_of(sA_u);

  v16bf Bcp[4], Bfu[8];
#pragma unroll
  for (int kc = 0; kc < 4; ++kc) Bcp[kc] = load_B_frag(Wcp, 32 * kc, n0, lane);
#pragma unroll
  for (int kc = 0; kc < 8; ++kc) Bfu[kc] = load_B_frag(Wfu, 32 * kc, n0, lane);
  const float biascp = bcp[nlane];
  const float biasfu = bfu[nlane];

  for (int tile = blockIdx.x; tile < n_tiles; tile += gridDim.x) {
    const int p0 = tile * 16;
    __syncthreads();
    {
      const int r = t >> 4, seg = t & 15;
      const int c0 = seg * 8;
      const int ci = p2c[p0 + r];
      // cluster_msg gather: bf16, async copy straight into LDS
      async_copy_b128(sA_base + (unsigned)(r * SH_STRIDE + c0) * 2u,
                      cmsg_bf + (size_t)ci * DFEAT + c0);
      // particle_msg: sequential fp32 rows, convert inline into combined left half
      const float4* pb = (const float4*)(pmsg + (size_t)(p0 + r) * DFEAT + c0);
      unsigned int* db = sC_u + (r * SA_STRIDE + c0) / 2;
#pragma unroll
      for (int q = 0; q < 2; ++q) {
        float4 vb = pb[q];
        db[2 * q]     = cvt2_bf16(vb.x, vb.y);
        db[2 * q + 1] = cvt2_bf16(vb.z, vb.w);
      }
    }
    wait_async0();
    __syncthreads();

    v8f c = {0.f, 0.f, 0.f, 0.f, 0.f, 0.f, 0.f, 0.f};
#pragma unroll
    for (int kc = 0; kc < 4; ++kc) {
      v16bf a = load_A_frag(sA, SH_STRIDE, 32 * kc, lane);
      c = wmma_bf16(a, Bcp[kc], c);
    }
#pragma unroll
    for (int r = 0; r < 8; ++r) {
      float h = fmaxf(c[r] + biascp, 0.0f);
      sC[(r + mhi) * SA_STRIDE + 128 + nlane] = f2bf16(h);
    }
    __syncthreads();

    v8f c2 = {0.f, 0.f, 0.f, 0.f, 0.f, 0.f, 0.f, 0.f};
#pragma unroll
    for (int kc = 0; kc < 8; ++kc) {
      v16bf a = load_A_frag(sC, SA_STRIDE, 32 * kc, lane);
      c2 = wmma_bf16(a, Bfu[kc], c2);
    }
#pragma unroll
    for (int r = 0; r < 8; ++r) {
      out[(size_t)(p0 + r + mhi) * DFEAT + nlane] = fmaxf(c2[r] + biasfu, 0.0f);
    }
  }
}

// ---------------------------------------------------------------------------
extern "C" void kernel_launch(void* const* d_in, const int* in_sizes, int n_in,
                              void* d_out, int out_size, void* d_ws, size_t ws_size,
                              hipStream_t stream) {
  const float* P      = (const float*)d_in[0];
  const int*   p2c    = (const int*)d_in[2];
  const int*   pedges = (const int*)d_in[3];
  const int*   cedges = (const int*)d_in[4];
  const float* W_li1 = (const float*)d_in[5];  const float* b_li1 = (const float*)d_in[6];
  const float* W_li2 = (const float*)d_in[7];  const float* b_li2 = (const float*)d_in[8];
  const float* W_pc  = (const float*)d_in[9];  const float* b_pc  = (const float*)d_in[10];
  const float* W_ci1 = (const float*)d_in[11]; const float* b_ci1 = (const float*)d_in[12];
  const float* W_ci2 = (const float*)d_in[13]; const float* b_ci2 = (const float*)d_in[14];
  const float* W_cp  = (const float*)d_in[15]; const float* b_cp  = (const float*)d_in[16];
  const float* W_fu  = (const float*)d_in[17]; const float* b_fu  = (const float*)d_in[18];

  const int N  = in_sizes[0] / DFEAT;
  const int C  = in_sizes[1] / DFEAT;
  const int EP = in_sizes[3] / 2;
  const int EC = in_sizes[4] / 2;

  char* ws = (char*)d_ws;
  size_t off = 0;
  auto alloc = [&](size_t bytes) -> void* {
    void* p = ws + off;
    off += (bytes + 255) & ~(size_t)255;
    return p;
  };
  float* particle_msg = (float*)alloc((size_t)N * DFEAT * sizeof(float));
  float* c_sum        = (float*)alloc((size_t)C * DFEAT * sizeof(float));
  float* counts       = (float*)alloc((size_t)C * sizeof(float));
  float* cluster_msg  = (float*)alloc((size_t)C * DFEAT * sizeof(float));
  unsigned short* Pbf     = (unsigned short*)alloc((size_t)N * DFEAT * 2);
  unsigned short* cagg_bf = (unsigned short*)alloc((size_t)C * DFEAT * 2);
  unsigned short* cmsg_bf = (unsigned short*)alloc((size_t)C * DFEAT * 2);
  unsigned short* wb_li1 = (unsigned short*)alloc((size_t)2 * DFEAT * DFEAT * 2);
  unsigned short* wb_li2 = (unsigned short*)alloc((size_t)DFEAT * DFEAT * 2);
  unsigned short* wb_pc  = (unsigned short*)alloc((size_t)DFEAT * DFEAT * 2);
  unsigned short* wb_ci1 = (unsigned short*)alloc((size_t)2 * DFEAT * DFEAT * 2);
  unsigned short* wb_ci2 = (unsigned short*)alloc((size_t)DFEAT * DFEAT * 2);
  unsigned short* wb_cp  = (unsigned short*)alloc((size_t)DFEAT * DFEAT * 2);
  unsigned short* wb_fu  = (unsigned short*)alloc((size_t)2 * DFEAT * DFEAT * 2);

  // Zero atomic accumulators (stream-ordered, graph-capturable).
  hipMemsetAsync(particle_msg, 0, (size_t)N * DFEAT * sizeof(float), stream);
  hipMemsetAsync(c_sum,        0, (size_t)C * DFEAT * sizeof(float), stream);
  hipMemsetAsync(counts,       0, (size_t)C * sizeof(float), stream);
  hipMemsetAsync(cluster_msg,  0, (size_t)C * DFEAT * sizeof(float), stream);

  // Weights + particle states -> bf16 tables.
  const int n2 = 2 * DFEAT * DFEAT, n1 = DFEAT * DFEAT;
  const int np = N * DFEAT;
  auto g4 = [](int n) { return (n / 4 + 255) / 256; };
  f32_to_bf16_kernel<<<g4(np), 256, 0, stream>>>(P, Pbf, np);
  f32_to_bf16_kernel<<<g4(n2), 256, 0, stream>>>(W_li1, wb_li1, n2);
  f32_to_bf16_kernel<<<g4(n1), 256, 0, stream>>>(W_li2, wb_li2, n1);
  f32_to_bf16_kernel<<<g4(n1), 256, 0, stream>>>(W_pc,  wb_pc,  n1);
  f32_to_bf16_kernel<<<g4(n2), 256, 0, stream>>>(W_ci1, wb_ci1, n2);
  f32_to_bf16_kernel<<<g4(n1), 256, 0, stream>>>(W_ci2, wb_ci2, n1);
  f32_to_bf16_kernel<<<g4(n1), 256, 0, stream>>>(W_cp,  wb_cp,  n1);
  f32_to_bf16_kernel<<<g4(n2), 256, 0, stream>>>(W_fu,  wb_fu,  n2);

  auto cap = [](int x) { return x < 8192 ? x : 8192; };

  // 1) Particle-edge MLP -> particle_msg (f32 atomic accum)
  const int ep_tiles = EP / 16;
  edge_mlp_kernel<<<cap(ep_tiles), 256, 0, stream>>>(
      Pbf, pedges, wb_li1, b_li1, wb_li2, b_li2, particle_msg, ep_tiles);

  // 2) Particle->cluster aggregation (sum + counts)
  const int p_tiles = N / 16;
  particle_cluster_kernel<<<cap(p_tiles), 256, 0, stream>>>(
      Pbf, p2c, wb_pc, b_pc, c_sum, counts, p_tiles);

  // 3) Mean -> bf16 cluster_agg table
  const int nc = C * DFEAT;
  mean_div_kernel<<<(nc + 255) / 256, 256, 0, stream>>>(c_sum, counts, cagg_bf, nc);

  // 4) Cluster-edge MLP -> cluster_msg (f32 atomic accum)
  const int ec_tiles = EC / 16;
  edge_mlp_kernel<<<cap(ec_tiles), 256, 0, stream>>>(
      cagg_bf, cedges, wb_ci1, b_ci1, wb_ci2, b_ci2, cluster_msg, ec_tiles);

  // 4b) cluster_msg -> bf16 gather table
  f32_to_bf16_kernel<<<g4(nc), 256, 0, stream>>>(cluster_msg, cmsg_bf, nc);

  // 5) Fused W_cp + concat + W_fu -> out
  fused_out_kernel<<<cap(p_tiles), 256, 0, stream>>>(
      particle_msg, cmsg_bf, p2c, wb_cp, b_cp, wb_fu, b_fu,
      (float*)d_out, p_tiles);
}
